// Head_63934883168755
// MI455X (gfx1250) — compile-verified, matrix operand order checked by reading, not run on hardware
//
#include <hip/hip_runtime.h>
#include <hip/hip_bf16.h>

// Problem constants (from reference)
#define B_ 16
#define T_ 2048
#define C_ 1024
#define H_ 128
#define SCALE_Q 0.08838834764831845f  // 1/sqrt(128)
#define NEG_BIG (-1.0e30f)

typedef __bf16 bf16;
typedef __attribute__((ext_vector_type(16))) __bf16 v16bf;
typedef __attribute__((ext_vector_type(8)))  __bf16 v8bf;
typedef __attribute__((ext_vector_type(8)))  float  v8f;
typedef __attribute__((ext_vector_type(4)))  float  v4f;

__device__ __forceinline__ v16bf cat8(v8bf lo, v8bf hi) {
    return __builtin_shufflevector(lo, hi, 0,1,2,3,4,5,6,7,8,9,10,11,12,13,14,15);
}
__device__ __forceinline__ v16bf load_frag(const bf16* p) {
    // 8 contiguous bf16 at p (16B) + 8 contiguous at p+16 (16B)
    v8bf lo = *(const v8bf*)(p);
    v8bf hi = *(const v8bf*)(p + 16);
    return cat8(lo, hi);
}
__device__ __forceinline__ v8f wmma_bf16(v16bf a, v16bf b, v8f c) {
    return __builtin_amdgcn_wmma_f32_16x16x32_bf16(false, a, false, b, (short)0, c, false, false);
}

// CDNA5 async global->LDS copy: each active lane moves 16 bytes.
// VDST = LDS byte address (low 32 bits of generic shared pointer),
// VADDR = 64-bit global address, saddr = off.  Tracked by ASYNCcnt.
__device__ __forceinline__ void async_copy_b128(unsigned lds_byte, const void* gaddr) {
    asm volatile("global_load_async_to_lds_b128 %0, %1, off"
                 :: "v"(lds_byte), "v"((unsigned long long)(uintptr_t)gaddr)
                 : "memory");
}
__device__ __forceinline__ void wait_asynccnt0()  { asm volatile("s_wait_asynccnt 0x0"  ::: "memory"); }
__device__ __forceinline__ void wait_asynccnt4()  { asm volatile("s_wait_asynccnt 0x4"  ::: "memory"); }
__device__ __forceinline__ void wait_asynccnt16() { asm volatile("s_wait_asynccnt 0x10" ::: "memory"); }

// ---------------------------------------------------------------------------
// Kernel 0: W [C][H] f32  ->  Wt [H][C] bf16   (for Wq, Wk, Wv)
// ---------------------------------------------------------------------------
__global__ void wt_transpose_kernel(const float* __restrict__ Wq,
                                    const float* __restrict__ Wk,
                                    const float* __restrict__ Wv,
                                    bf16* __restrict__ Wt) {
    int idx = blockIdx.x * blockDim.x + threadIdx.x;      // [0, 3*C*H)
    int mat = idx / (C_ * H_);
    int r   = idx % (C_ * H_);
    int h   = r / C_;
    int c   = r % C_;
    const float* W = (mat == 0) ? Wq : (mat == 1) ? Wk : Wv;
    Wt[(size_t)mat * H_ * C_ + (size_t)h * C_ + c] = (bf16)W[(size_t)c * H_ + h];
}

// ---------------------------------------------------------------------------
// Kernel 1: QKV projection.  Q[b][t][h]*scale, K[b][t][h], Vt[b][h][t] (bf16)
// Block = 4 waves; each wave owns a 16-row strip; all waves SHARE the weight
// chunk.  Per k-step the 8KB W chunk (128x32 bf16) is staged into LDS with
// async b128 copies, double-buffered one iteration ahead, so B-fragments are
// short-latency ds_load_b128 instead of L2 round trips before every WMMA.
// ---------------------------------------------------------------------------
__global__ void qkv_proj_kernel(const float* __restrict__ x,
                                const bf16* __restrict__ Wt,   // [3][H][C]
                                bf16* __restrict__ Q,          // [B][T][H]
                                bf16* __restrict__ K,          // [B][T][H]
                                bf16* __restrict__ Vt) {       // [B][H][T]
    const int tid   = threadIdx.x;
    const int lane  = tid & 31;
    const int wave  = tid >> 5;
    const int tile  = blockIdx.x * 4 + wave;        // global 16-row tile
    const int b     = tile / (T_ / 16);
    const int r0    = (tile % (T_ / 16)) * 16;
    const int ln    = lane & 15;
    const int khalf = (lane >> 4) * 8;

    __shared__ __align__(16) bf16 Wbuf[2][128 * 32];   // double-buffered W chunk (16KB)

    const float* xrow = x + ((size_t)b * T_ + r0 + ln) * C_;

    // Each of the 128 threads copies 4 x 16B of the 8KB chunk.
    auto issue_wchunk = [&](const bf16* Wm, int kc, int bufidx) {
        const char* g = (const char*)Wm + (size_t)kc * 64;   // col byte offset in each row
        unsigned base = (unsigned)(uintptr_t)&Wbuf[bufidx][0];
        #pragma unroll
        for (int it = 0; it < 4; ++it) {
            unsigned flat = (unsigned)it * 2048u + (unsigned)tid * 16u; // byte in chunk
            unsigned row  = flat >> 6;        // h row (64B per row)
            unsigned cb   = flat & 63u;       // byte within row
            async_copy_b128(base + flat, g + (size_t)row * (C_ * 2) + cb);
        }
    };

    #pragma unroll
    for (int mat = 0; mat < 3; ++mat) {
        const bf16* Wm = Wt + (size_t)mat * H_ * C_;
        v8f acc[8];
        #pragma unroll
        for (int nt = 0; nt < 8; ++nt) acc[nt] = (v8f)0.0f;

        issue_wchunk(Wm, 0, 0);                        // prologue

        for (int kc = 0; kc < C_ / 32; ++kc) {
            const bool have_next = (kc + 1) < (C_ / 32);
            if (have_next) issue_wchunk(Wm, kc + 1, (kc + 1) & 1);

            // A fragment from x (f32 -> bf16); global loads overlap the waits.
            const int c0 = kc * 32 + khalf;
            v4f f0 = *(const v4f*)(xrow + c0);
            v4f f1 = *(const v4f*)(xrow + c0 + 4);
            v4f f2 = *(const v4f*)(xrow + c0 + 16);
            v4f f3 = *(const v4f*)(xrow + c0 + 20);

            if (have_next) wait_asynccnt4();           // own chunk-kc parts landed
            else           wait_asynccnt0();
            __syncthreads();                           // all waves' parts landed

            v16bf afrag;
            #pragma unroll
            for (int t = 0; t < 4; ++t) {
                afrag[t]      = (bf16)f0[t];
                afrag[4 + t]  = (bf16)f1[t];
                afrag[8 + t]  = (bf16)f2[t];
                afrag[12 + t] = (bf16)f3[t];
            }
            const bf16* wb = &Wbuf[kc & 1][0];
            #pragma unroll
            for (int nt = 0; nt < 8; ++nt) {
                v16bf bfrag = load_frag(wb + (nt * 16 + ln) * 32 + khalf);
                acc[nt] = wmma_bf16(afrag, bfrag, acc[nt]);
            }
            __syncthreads();                           // buffer may be refilled next iter
        }

        // Store: C/D layout -> n = lane%16, m = i + 8*(lane/16)
        #pragma unroll
        for (int nt = 0; nt < 8; ++nt) {
            #pragma unroll
            for (int i = 0; i < 8; ++i) {
                const int row = r0 + i + 8 * (lane >> 4);
                const int h   = nt * 16 + ln;
                float v = acc[nt][i];
                if (mat == 0) {
                    Q[((size_t)b * T_ + row) * H_ + h] = (bf16)(v * SCALE_Q);
                } else if (mat == 1) {
                    K[((size_t)b * T_ + row) * H_ + h] = (bf16)v;
                } else {
                    Vt[((size_t)b * H_ + h) * T_ + row] = (bf16)v;
                }
            }
        }
    }
}

// ---------------------------------------------------------------------------
// Kernel 2: causal flash attention.  One wave owns one 16-query tile.
// Block = 8 waves (256 threads).  Streams 32-key tiles with online softmax.
// V tiles (128 x 32 bf16 = 8KB) are double-buffered in LDS via
// global_load_async_to_lds_b128; next K tile is warmed with global_prefetch.
// ---------------------------------------------------------------------------
__global__ void attn_kernel(const bf16* __restrict__ Q,   // [B][T][H] (pre-scaled)
                            const bf16* __restrict__ K,   // [B][T][H]
                            const bf16* __restrict__ Vt,  // [B][H][T]
                            float* __restrict__ out) {    // [B][T][H]
    const int lane  = threadIdx.x & 31;
    const int wave  = threadIdx.x >> 5;
    const int ln    = lane & 15;
    const int khalf = (lane >> 4) * 8;

    const int blocksPerBatch = (T_ / 16) / 8;             // 16
    const int b  = blockIdx.x / blocksPerBatch;
    const int qt = (blockIdx.x % blocksPerBatch) * 8 + wave;
    const int r0 = qt * 16;

    __shared__ __align__(16) bf16 Pbuf[8][16][32];           // per-wave P staging (8KB)
    __shared__ __align__(16) bf16 Vbuf[8][2][128 * 32];      // per-wave V tiles (128KB)

    // Load the 16x128 query strip as 4 A-fragments.
    v16bf aq[4];
    {
        const bf16* qrow = Q + ((size_t)b * T_ + r0 + ln) * H_;
        #pragma unroll
        for (int hc = 0; hc < 4; ++hc)
            aq[hc] = load_frag(qrow + hc * 32 + khalf);
    }

    float mrow[8], lrow[8];
    v8f accO[8];
    #pragma unroll
    for (int i = 0; i < 8; ++i) { mrow[i] = NEG_BIG; lrow[i] = 0.0f; }
    #pragma unroll
    for (int nt = 0; nt < 8; ++nt) accO[nt] = (v8f)0.0f;

    const bf16* Kb  = K  + (size_t)b * T_ * H_;
    const char* Vtb = (const char*)(Vt + (size_t)b * H_ * T_);
    const unsigned vbase0 = (unsigned)(uintptr_t)&Vbuf[wave][0][0];  // LDS byte offset
    const unsigned vbase1 = (unsigned)(uintptr_t)&Vbuf[wave][1][0];
    const int ktiles = (r0 + 16 + 31) >> 5;

    // Issue the async copy of one 8KB V tile [h][key-in-tile] (64B rows).
    auto issue_vtile = [&](int k0, unsigned vbase) {
        const char* gv = Vtb + (size_t)k0 * 2;
        #pragma unroll
        for (int it = 0; it < 16; ++it) {
            unsigned flat = (unsigned)it * 512u + (unsigned)lane * 16u; // byte in tile
            unsigned row  = flat >> 6;       // h
            unsigned cb   = flat & 63u;      // byte within row
            async_copy_b128(vbase + flat, gv + ((size_t)row << 12) + cb);
        }
    };

    issue_vtile(0, vbase0);   // prologue: tile 0 -> buffer 0

    for (int kt = 0; kt < ktiles; ++kt) {
        const int k0   = kt * 32;
        const int key0 = k0 + ln;
        const bool have_next = (kt + 1) < ktiles;

        // ---- kick off NEXT tile's async V copy; warm next K tile in L2/L0 ----
        if (have_next) {
            issue_vtile(k0 + 32, (kt & 1) ? vbase0 : vbase1);
            const bf16* nk = Kb + (size_t)(key0 + 32) * H_;
            __builtin_prefetch(nk, 0, 3);
            __builtin_prefetch(nk + 16 * H_, 0, 3);
        }

        // ---- S = Q K^T (two 16x16 score tiles, K-dim = H = 128) ----
        v16bf bk[8];
        {
            const bf16* krow0 = Kb + (size_t)key0 * H_;
            const bf16* krow1 = Kb + (size_t)(key0 + 16) * H_;
            #pragma unroll
            for (int hc = 0; hc < 4; ++hc) {
                const int c = hc * 32 + khalf;
                bk[hc]     = load_frag(krow0 + c);
                bk[4 + hc] = load_frag(krow1 + c);
            }
        }
        v8f s0 = (v8f)0.0f, s1 = (v8f)0.0f;
        #pragma unroll
        for (int hc = 0; hc < 4; ++hc) {
            s0 = wmma_bf16(aq[hc], bk[hc],     s0);
            s1 = wmma_bf16(aq[hc], bk[4 + hc], s1);
        }

        // ---- causal mask + online softmax (row = i + 8*(lane/16)) ----
        float alpha[8];
        #pragma unroll
        for (int i = 0; i < 8; ++i) {
            const int row = r0 + i + 8 * (lane >> 4);
            float v0 = (key0      <= row) ? s0[i] : NEG_BIG;
            float v1 = (key0 + 16 <= row) ? s1[i] : NEG_BIG;
            float mx = fmaxf(v0, v1);
            #pragma unroll
            for (int off = 1; off < 16; off <<= 1)
                mx = fmaxf(mx, __shfl_xor(mx, off, 32));  // reduce over 16-lane half
            const float mnew = fmaxf(mrow[i], mx);
            alpha[i] = __expf(mrow[i] - mnew);
            mrow[i]  = mnew;
            const float p0 = __expf(v0 - mnew);
            const float p1 = __expf(v1 - mnew);
            float rs = p0 + p1;
            #pragma unroll
            for (int off = 1; off < 16; off <<= 1)
                rs += __shfl_xor(rs, off, 32);
            lrow[i] = lrow[i] * alpha[i] + rs;
            // Stage P in A-matrix-friendly LDS layout: Pbuf[m][key-in-tile]
            const int m = i + 8 * (lane >> 4);
            Pbuf[wave][m][ln]      = (bf16)p0;
            Pbuf[wave][m][16 + ln] = (bf16)p1;
        }
        #pragma unroll
        for (int nt = 0; nt < 8; ++nt) {
            v8f t = accO[nt];
            #pragma unroll
            for (int i = 0; i < 8; ++i) t[i] *= alpha[i];
            accO[nt] = t;
        }

        // Re-read P as an A fragment (same-wave LDS; compiler inserts dscnt waits)
        v16bf ap = load_frag(&Pbuf[wave][ln][khalf]);

        // ---- O += P V : V fragments come from the async-filled LDS tile ----
        if (have_next) wait_asynccnt16();   // tile kt landed; kt+1 still in flight
        else           wait_asynccnt0();
        #pragma unroll
        for (int nt = 0; nt < 8; ++nt) {
            const bf16* vp = &Vbuf[wave][kt & 1][(nt * 16 + ln) * 32 + khalf];
            accO[nt] = wmma_bf16(ap, load_frag(vp), accO[nt]);
        }
    }

    // ---- epilogue: normalize and store f32 ----
    #pragma unroll
    for (int nt = 0; nt < 8; ++nt) {
        #pragma unroll
        for (int i = 0; i < 8; ++i) {
            const int row = r0 + i + 8 * (lane >> 4);
            const int h   = nt * 16 + ln;
            out[((size_t)b * T_ + row) * H_ + h] = accO[nt][i] / lrow[i];
        }
    }
}

// ---------------------------------------------------------------------------
extern "C" void kernel_launch(void* const* d_in, const int* in_sizes, int n_in,
                              void* d_out, int out_size, void* d_ws, size_t ws_size,
                              hipStream_t stream) {
    const float* x  = (const float*)d_in[0];
    const float* Wq = (const float*)d_in[1];
    const float* Wk = (const float*)d_in[2];
    const float* Wv = (const float*)d_in[3];
    float* out = (float*)d_out;

    // Workspace carve-up (bf16)
    char* ws = (char*)d_ws;
    const size_t WT_BYTES  = (size_t)3 * H_ * C_ * sizeof(bf16);   // 768 KB
    const size_t QKV_BYTES = (size_t)B_ * T_ * H_ * sizeof(bf16);  // 8 MB each
    bf16* Wt = (bf16*)(ws);
    bf16* Q  = (bf16*)(ws + WT_BYTES);
    bf16* K  = (bf16*)(ws + WT_BYTES + QKV_BYTES);
    bf16* Vt = (bf16*)(ws + WT_BYTES + 2 * QKV_BYTES);

    // 0) weight transpose + convert
    {
        int total = 3 * C_ * H_;
        wt_transpose_kernel<<<total / 256, 256, 0, stream>>>(Wq, Wk, Wv, Wt);
    }
    // 1) QKV projection (4 waves/block, LDS-staged weights)
    {
        int tiles = B_ * (T_ / 16);            // 2048
        qkv_proj_kernel<<<tiles / 4, 128, 0, stream>>>(x, Wt, Q, K, Vt);
    }
    // 2) causal flash attention (8 waves/block, 16 queries per wave)
    {
        int tiles = B_ * (T_ / 16);            // 2048
        attn_kernel<<<tiles / 8, 256, 0, stream>>>(Q, K, Vt, out);
    }
}
